// OccupancyDecoder_14499809592081
// MI455X (gfx1250) — matrix operand order repaired and checked by reading, not compiled, not run on hardware
//
#include <hip/hip_runtime.h>
#include <hip/hip_bf16.h>

#define NANCH 4096
#define NVOX  32768
#define HIDN  128
// output strides for occ[B][128][128][128][16]
#define OSTR_B 33554432
#define OSTR_J 262144
#define ROW_STRIDE 40   // halves per LDS row (80B: 16B-aligned, bank-spread)

typedef __attribute__((ext_vector_type(16))) __bf16 v16bf;
typedef __attribute__((ext_vector_type(8)))  __bf16 v8bf;
typedef __attribute__((ext_vector_type(8)))  float  v8f;

__device__ __forceinline__ float anchor_pos(int i) {
    return -50.0f + (100.0f / 15.0f) * (float)i;
}

// ---------------- MLP: x = relu(in @ W1 + b1) @ W2 + b2, stored transposed bf16 ----
__global__ void mlp_kernel(const float* __restrict__ pos, const float* __restrict__ scl,
                           const float* __restrict__ rot, const float* __restrict__ opa,
                           const float* __restrict__ W1, const float* __restrict__ b1,
                           const float* __restrict__ W2, const float* __restrict__ b2,
                           __bf16* __restrict__ xT) {
    const int v = blockIdx.x;            // 0 .. B*N-1
    const int t = threadIdx.x;           // 0 .. 127
    const int b = v >> 12;
    const int n = v & (NANCH - 1);
    __shared__ float h[HIDN];
    float in[11];
#pragma unroll
    for (int c = 0; c < 3; ++c) in[c]     = pos[v * 3 + c];
#pragma unroll
    for (int c = 0; c < 3; ++c) in[3 + c] = scl[v * 3 + c];
#pragma unroll
    for (int c = 0; c < 4; ++c) in[6 + c] = rot[v * 4 + c];
    in[10] = opa[v];
    float acc = b1[t];
#pragma unroll
    for (int c = 0; c < 11; ++c) acc = fmaf(in[c], W1[c * HIDN + t], acc);
    h[t] = fmaxf(acc, 0.0f);
    __syncthreads();
    float o = b2[t];
#pragma unroll 8
    for (int k = 0; k < HIDN; ++k) o = fmaf(h[k], W2[k * HIDN + t], o);
    xT[(size_t)b * HIDN * NANCH + (size_t)t * NANCH + n] = (__bf16)o;
}

// ---------------- Deterministic partition of voxels by batch (stable order) --------
__global__ void partition_kernel(const int* __restrict__ vcoord,
                                 int* __restrict__ perm0, int* __restrict__ perm1,
                                 int* __restrict__ hdr) {
    __shared__ int scan[1024];
    const int tid  = threadIdx.x;
    const int base = tid * 32;
    unsigned mask = 0u;
    int cnt = 0;
#pragma unroll 4
    for (int k = 0; k < 32; ++k) {
        int bb = vcoord[(base + k) * 4];
        if (bb == 0) { mask |= (1u << k); ++cnt; }
    }
    scan[tid] = cnt;
    __syncthreads();
    for (int off = 1; off < 1024; off <<= 1) {
        int vv = scan[tid];
        int uu = (tid >= off) ? scan[tid - off] : 0;
        __syncthreads();
        scan[tid] = vv + uu;
        __syncthreads();
    }
    const int incl   = scan[tid];
    const int total0 = scan[1023];
    int o0 = incl - cnt;
    int o1 = base - o0;
    for (int k = 0; k < 32; ++k) {
        int idx = base + k;
        if (mask & (1u << k)) perm0[o0++] = idx;
        else                  perm1[o1++] = idx;
    }
    const int total1 = NVOX - total0;
    const int t0p = (total0 + 15) & ~15;
    const int t1p = (total1 + 15) & ~15;
    if (tid < 16) {
        if (total0 + tid < t0p) perm0[total0 + tid] = -1;
        if (total1 + tid < t1p) perm1[total1 + tid] = -1;
    }
    if (tid == 0) { hdr[0] = total0; hdr[1] = total1; hdr[2] = t0p >> 4; hdr[3] = t1p >> 4; }
}

// ---------------- Zero output ------------------------------------------------------
__global__ void zero_kernel(float4* __restrict__ out, int n4) {
    int i = blockIdx.x * blockDim.x + threadIdx.x;
    const int stride = gridDim.x * blockDim.x;
    for (; i < n4; i += stride) out[i] = make_float4(0.f, 0.f, 0.f, 0.f);
}

// ---------------- Fused dist+softmax+GEMM+scatter ----------------------------------
// 8 waves/block, one 16-voxel tile per wave, all tiles in a block share a batch.
// B chunks (32 anchors x 128 hid, bf16) are double-buffered in LDS via
// global_load_async_to_lds_b128 and shared by all 8 waves.
__global__ void __launch_bounds__(256) fused_kernel(
        const __bf16* __restrict__ xT, const int* __restrict__ vcoord,
        const int* __restrict__ perm0, const int* __restrict__ perm1,
        const int* __restrict__ hdr, float* __restrict__ out) {
    __shared__ __bf16 sbuf[2][128 * ROW_STRIDE];   // 2 x 10KB

    const int tiles0  = hdr[2];
    const int tiles1  = hdr[3];
    const int blocks0 = (tiles0 + 7) >> 3;
    const int blocks1 = (tiles1 + 7) >> 3;
    if ((int)blockIdx.x >= blocks0 + blocks1) return;   // block-uniform exit

    int b, tilebase, tiles;
    const int* perm;
    if ((int)blockIdx.x < blocks0) { b = 0; perm = perm0; tilebase = blockIdx.x * 8;             tiles = tiles0; }
    else                           { b = 1; perm = perm1; tilebase = (blockIdx.x - blocks0) * 8; tiles = tiles1; }

    const int lane = threadIdx.x & 31;
    const int wv   = threadIdx.x >> 5;              // wave in block: 0..7
    int tile = tilebase + wv;
    const int active = (tile < tiles) ? 1 : 0;
    if (!active) tile = tiles - 1;                  // tiles >= 1 guaranteed here

    const int m  = lane & 15;                       // A row / B column within tile
    const int hi = lane >> 4;                       // lane-half select
    int vIdx = perm[tile * 16 + m];
    const int valid = (active && vIdx >= 0) ? 1 : 0;
    if (vIdx < 0) vIdx = 0;

    const int4 vc = ((const int4*)vcoord)[vIdx];    // (b, x, y, z)
    const float fx = (float)vc.y, fy = (float)vc.z, fz = (float)vc.w;
    const int vbase = vc.x * OSTR_B + vc.y * 2048 + vc.z * 16 + vc.w;

    // dmin = sqrt(min_i dx2 + min_j dy2 + min_k dz2)  (separable grid)
    float mnx = 1e30f, mny = 1e30f, mnz = 1e30f;
#pragma unroll
    for (int i = 0; i < 16; ++i) {
        float a = anchor_pos(i);
        float dx = fx - a; mnx = fminf(mnx, dx * dx);
        float dy = fy - a; mny = fminf(mny, dy * dy);
        float dz = fz - a; mnz = fminf(mnz, dz * dz);
    }
    const float dmin = __builtin_amdgcn_sqrtf(mnx + mny + mnz);

    // per-lane fixed z-axis table (A layout: this lane covers kz = 8*hi .. 8*hi+7)
    float dz2[8];
    const int kzb = hi * 8;
#pragma unroll
    for (int t = 0; t < 8; ++t) { float d = fz - anchor_pos(kzb + t); dz2[t] = d * d; }

    v8f cacc[8];
#pragma unroll
    for (int jt = 0; jt < 8; ++jt) cacc[jt] = (v8f)0.0f;
    float rs = 0.0f;

    // staging assignment: thread stages 32B = row (tid>>1), halves 16*(tid&1)..+15
    const __bf16* xb  = xT + (size_t)b * (HIDN * NANCH);
    const int srow    = threadIdx.x >> 1;
    const int shalf   = (threadIdx.x & 1) << 4;
    const __bf16* src0 = xb + (size_t)srow * NANCH + shalf;
    const unsigned lds0 = (unsigned)(uintptr_t)(&sbuf[0][0] + srow * ROW_STRIDE + shalf);
    const unsigned lds1 = (unsigned)(uintptr_t)(&sbuf[1][0] + srow * ROW_STRIDE + shalf);

    // stage chunk 0 into buffer 0
    {
        unsigned long long g = (unsigned long long)(uintptr_t)src0;
        asm volatile("global_load_async_to_lds_b128 %0, %1, off\n\t"
                     "global_load_async_to_lds_b128 %0, %1, off offset:16"
                     :: "v"(lds0), "v"(g) : "memory");
    }
    asm volatile("s_wait_asynccnt 0x0" ::: "memory");
    __syncthreads();

    const float LOG2E = 1.4426950408889634f;

    for (int c = 0; c < 128; ++c) {                 // K chunks of 32 anchors
        // prefetch next chunk into the other buffer (async, overlaps compute)
        if (c + 1 < 128) {
            unsigned long long g = (unsigned long long)(uintptr_t)(src0 + ((c + 1) << 5));
            unsigned ldsn = ((c + 1) & 1) ? lds1 : lds0;
            asm volatile("global_load_async_to_lds_b128 %0, %1, off\n\t"
                         "global_load_async_to_lds_b128 %0, %1, off offset:16"
                         :: "v"(ldsn), "v"(g) : "memory");
        }

        const int j0 = (c & 7) << 1;
        float dxv = fx - anchor_pos(c >> 3);
        float dx2 = dxv * dxv;
        float dy0 = fy - anchor_pos(j0);     float s0 = dx2 + dy0 * dy0;
        float dy1 = fy - anchor_pos(j0 + 1); float s1 = dx2 + dy1 * dy1;

        v16bf a;                                    // A frag: softmax numerators, bf16
#pragma unroll
        for (int t = 0; t < 8; ++t) {
            float d0 = __builtin_amdgcn_sqrtf(s0 + dz2[t]);
            float d1 = __builtin_amdgcn_sqrtf(s1 + dz2[t]);
            float w0 = __builtin_amdgcn_exp2f((dmin - d0) * LOG2E);
            float w1 = __builtin_amdgcn_exp2f((dmin - d1) * LOG2E);
            rs += w0 + w1;
            a[t]     = (__bf16)w0;                  // e=t   -> K = 8*hi + t
            a[t + 8] = (__bf16)w1;                  // e=t+8 -> K = 16 + 8*hi + t
        }

        const __bf16* cb = &sbuf[c & 1][0];
        const __bf16* fp = cb + (size_t)m * ROW_STRIDE + (hi << 3);
#pragma unroll
        for (int jt = 0; jt < 8; ++jt) {
            const __bf16* p = fp + (size_t)jt * (16 * ROW_STRIDE);
            v8bf lo = *(const v8bf*)(p);
            v8bf hg = *(const v8bf*)(p + 16);
            v16bf bb = __builtin_shufflevector(lo, hg, 0, 1, 2, 3, 4, 5, 6, 7,
                                                       8, 9, 10, 11, 12, 13, 14, 15);
            cacc[jt] = __builtin_amdgcn_wmma_f32_16x16x32_bf16(
                false, a, false, bb, (short)0, cacc[jt], false, false);
        }

        asm volatile("s_wait_asynccnt 0x0" ::: "memory");
        __syncthreads();
    }

    // softmax denominator: combine the two K-halves of each row
    float rsf = rs + __shfl_xor(rs, 16, 32);
    float inv = 1.0f / rsf;                         // exp(0) term guarantees rsf >= 1

    // scatter: C layout -> lane holds column n=m, rows r + 8*hi
#pragma unroll
    for (int r = 0; r < 8; ++r) {
        const int src = r + (hi << 3);
        float invr = __shfl(inv, src, 32);
        int   vbr  = __shfl(vbase, src, 32);
        int   okr  = __shfl(valid, src, 32);
        if (okr) {
#pragma unroll
            for (int jt = 0; jt < 8; ++jt) {
                out[(size_t)vbr + (size_t)((jt << 4) + m) * OSTR_J] = cacc[jt][r] * invr;
            }
        }
    }
}

extern "C" void kernel_launch(void* const* d_in, const int* in_sizes, int n_in,
                              void* d_out, int out_size, void* d_ws, size_t ws_size,
                              hipStream_t stream) {
    const float* pos = (const float*)d_in[0];
    const float* scl = (const float*)d_in[1];
    const float* rot = (const float*)d_in[2];
    const float* opa = (const float*)d_in[3];
    const float* W1  = (const float*)d_in[4];
    const float* b1  = (const float*)d_in[5];
    const float* W2  = (const float*)d_in[6];
    const float* b2  = (const float*)d_in[7];
    const int* vcoord = (const int*)d_in[8];
    float* out = (float*)d_out;

    // workspace layout
    char* ws = (char*)d_ws;
    int* hdr   = (int*)ws;                                   // 16 ints
    int* perm0 = (int*)(ws + 64);                            // NVOX+16 ints
    int* perm1 = perm0 + (NVOX + 16);
    __bf16* xT = (__bf16*)(ws + 64 + 2 * (size_t)(NVOX + 16) * sizeof(int)); // 2MB, 64B aligned

    zero_kernel<<<4096, 256, 0, stream>>>((float4*)out, out_size / 4);
    mlp_kernel<<<2 * NANCH, 128, 0, stream>>>(pos, scl, rot, opa, W1, b1, W2, b2, xT);
    partition_kernel<<<1, 1024, 0, stream>>>(vcoord, perm0, perm1, hdr);

    // ceil(tiles0/8)+ceil(tiles1/8) <= 258 blocks of 8 waves
    fused_kernel<<<258, 256, 0, stream>>>(xT, vcoord, perm0, perm1, hdr, out);
}
